// MMDLoss_6090263625755
// MI455X (gfx1250) — matrix operand order repaired and checked by reading, not compiled, not run on hardware
//
#include <hip/hip_runtime.h>

typedef _Float16 h8   __attribute__((ext_vector_type(8)));
typedef _Float16 v16h __attribute__((ext_vector_type(16)));
typedef float    v8f  __attribute__((ext_vector_type(8)));
typedef int      v4i  __attribute__((ext_vector_type(4)));

#define N_TOTAL 8192
#define BHALF   4096
#define DDIM    256
#define LROW    264          // padded LDS row stride (halves): 528B -> bank-conflict-free
#define NTILE   128          // 64-wide block tiles per matrix dim
#define NPAIRS  8256         // NTILE*(NTILE+1)/2 upper-triangular block pairs

// workspace byte offsets
#define XH_OFF   0u          // 8192*256 halves = 4,194,304 B
#define SQ_OFF   4194304u    // 8192 f32
#define COL_OFF  4227072u    // 256 blocks * 256 f32 column-sum partials
#define SSQ_OFF  4489216u    // 256 f32 sum-of-squares partials
#define COEF_OFF 4490240u    // 1 f32
#define PART_OFF 4490496u    // 8256 f32 gemm block partials

// ---------------------------------------------------------------------------
// K1: row squared norms, f32->f16 conversion, per-block colsum / sumsq partials
// grid 256 blocks x 256 threads; block handles 32 rows (8 waves x 4 rows)
// ---------------------------------------------------------------------------
__global__ void __launch_bounds__(256)
mmd_prep_kernel(const float* __restrict__ src, const float* __restrict__ tgt,
                char* __restrict__ ws) {
  _Float16* Xh  = (_Float16*)(ws + XH_OFF);
  float* sq     = (float*)(ws + SQ_OFF);
  float* colp   = (float*)(ws + COL_OFF);
  float* ssqp   = (float*)(ws + SSQ_OFF);

  __shared__ float cshare[8][256];
  __shared__ float wshare[8];

  const int t  = threadIdx.x;
  const int w  = t >> 5;          // wave id (wave32)
  const int l  = t & 31;          // lane id
  const int r0 = blockIdx.x * 32 + w * 4;

  float csum[8];
#pragma unroll
  for (int k = 0; k < 8; ++k) csum[k] = 0.f;
  float wssq = 0.f;

  for (int rr = 0; rr < 4; ++rr) {
    const int row = r0 + rr;
    const float* rp = (row < BHALF) ? (src + (size_t)row * DDIM)
                                    : (tgt + (size_t)(row - BHALF) * DDIM);
    float4 v0 = ((const float4*)rp)[l * 2 + 0];
    float4 v1 = ((const float4*)rp)[l * 2 + 1];
    float vv[8] = {v0.x, v0.y, v0.z, v0.w, v1.x, v1.y, v1.z, v1.w};
    float p = 0.f;
    h8 hv;
#pragma unroll
    for (int k = 0; k < 8; ++k) {
      float v = vv[k];
      p += v * v;
      csum[k] += v;
      hv[k] = (_Float16)v;
    }
    *((h8*)(Xh + (size_t)row * DDIM + l * 8)) = hv;
    // wave32 reduction of row squared norm
#pragma unroll
    for (int off = 16; off > 0; off >>= 1) p += __shfl_down(p, off);
    if (l == 0) { sq[row] = p; wssq += p; }
  }
#pragma unroll
  for (int k = 0; k < 8; ++k) cshare[w][l * 8 + k] = csum[k];
  if (l == 0) wshare[w] = wssq;
  __syncthreads();
  float cp = 0.f;
#pragma unroll
  for (int wi = 0; wi < 8; ++wi) cp += cshare[wi][t];   // fixed order: deterministic
  colp[blockIdx.x * 256 + t] = cp;
  if (t == 0) {
    float s = 0.f;
    for (int wi = 0; wi < 8; ++wi) s += wshare[wi];
    ssqp[blockIdx.x] = s;
  }
}

// ---------------------------------------------------------------------------
// K2: closed-form bandwidth: sum(l2) = 2*N*sumsq - 2*||colsum||^2
// coef = -4 / bandwidth   (smallest kernel denominator = bw/4)
// ---------------------------------------------------------------------------
__global__ void __launch_bounds__(256)
mmd_bw_kernel(char* __restrict__ ws) {
  const float* colp = (const float*)(ws + COL_OFF);
  const float* ssqp = (const float*)(ws + SSQ_OFF);
  float* coef       = (float*)(ws + COEF_OFF);
  __shared__ float red[256];
  const int t = threadIdx.x;

  float s = 0.f;
  for (int b2 = 0; b2 < 256; ++b2) s += colp[b2 * 256 + t];
  red[t] = s * s;
  __syncthreads();
  for (int st = 128; st > 0; st >>= 1) {
    if (t < st) red[t] += red[t + st];
    __syncthreads();
  }
  const float ss = red[0];
  __syncthreads();
  red[t] = ssqp[t];
  __syncthreads();
  for (int st = 128; st > 0; st >>= 1) {
    if (t < st) red[t] += red[t + st];
    __syncthreads();
  }
  if (t == 0) {
    double sumsq  = (double)red[0];
    double sum_l2 = 2.0 * (double)N_TOTAL * sumsq - 2.0 * (double)ss;
    double bw     = sum_l2 / ((double)N_TOTAL * (double)N_TOTAL - (double)N_TOTAL);
    coef[0] = (float)(-4.0 / bw);
  }
}

// ---------------------------------------------------------------------------
// K3: fused X*X^T tile + MMD kernel epilogue, upper-triangular block tiles only.
// grid NPAIRS x 128 threads (4 waves). Block tile 64x64 at (bi,bj), bi<=bj.
// Panels staged in LDS (async-to-LDS when available), WMMAs fed by ds_load_b128.
// ---------------------------------------------------------------------------
__global__ void __launch_bounds__(128)
mmd_gemm_kernel(char* __restrict__ ws) {
  const _Float16* Xh = (const _Float16*)(ws + XH_OFF);
  const float* sq    = (const float*)(ws + SQ_OFF);
  const float coef   = *((const float*)(ws + COEF_OFF));
  float* part        = (float*)(ws + PART_OFF);

  __shared__ __align__(16) _Float16 As[64 * LROW];
  __shared__ __align__(16) _Float16 Bs[64 * LROW];

  const int t    = threadIdx.x;
  const int w    = t >> 5;
  const int lane = t & 31;
  const int l15  = lane & 15;
  const int hi   = (lane >> 4) & 1;     // lane group 16..31

  // Triangular decode: blockIdx.x -> (bi, bj) with bi <= bj.
  // S(bi) = bi*NTILE - bi*(bi-1)/2 is the first linear index of row bi.
  const int p = (int)blockIdx.x;
  int bi = (int)((2.f * NTILE + 1.f -
                  sqrtf((float)((2 * NTILE + 1) * (2 * NTILE + 1) - 8 * p))) * 0.5f);
  if (bi < 0) bi = 0;
  if (bi > NTILE - 1) bi = NTILE - 1;
  while (bi > 0 && (bi * NTILE - (bi * (bi - 1)) / 2) > p) --bi;
  while (((bi + 1) * NTILE - ((bi + 1) * bi) / 2) <= p) ++bi;
  const int bj = bi + (p - (bi * NTILE - (bi * (bi - 1)) / 2));
  const bool diagBlk = (bi == bj);

  const int ibase = bi * 64 + w * 16;
  const int jbase = bj * 64;

  // ---- Stage A (rows bi*64..+63) and B (rows bj*64..+63) panels into LDS ----
  // 64 rows x 32 16B-chunks = 2048 chunks per panel; 16 chunks per thread.
  {
    const _Float16* gA = Xh + (size_t)bi * 64 * DDIM;
    const _Float16* gB = Xh + (size_t)bj * 64 * DDIM;
#if defined(__gfx1250__) && __has_builtin(__builtin_amdgcn_global_load_async_to_lds_b128)
    for (int c = 0; c < 16; ++c) {
      const int chunk = c * 128 + t;
      const int row   = chunk >> 5;
      const int col8  = (chunk & 31) * 8;
      const int goff  = row * DDIM + col8;
      const int loff  = row * LROW + col8;
      __builtin_amdgcn_global_load_async_to_lds_b128(
          (v4i*)(gA + goff), (v4i*)(As + loff), 0, 0);
      if (!diagBlk)
        __builtin_amdgcn_global_load_async_to_lds_b128(
            (v4i*)(gB + goff), (v4i*)(Bs + loff), 0, 0);
    }
#if __has_builtin(__builtin_amdgcn_s_wait_asynccnt)
    __builtin_amdgcn_s_wait_asynccnt(0);
#else
    asm volatile("s_wait_asynccnt 0x0" ::: "memory");
#endif
#else
    for (int c = 0; c < 16; ++c) {
      const int chunk = c * 128 + t;
      const int row   = chunk >> 5;
      const int col8  = (chunk & 31) * 8;
      *(h8*)(As + row * LROW + col8) = *(const h8*)(gA + row * DDIM + col8);
      if (!diagBlk)
        *(h8*)(Bs + row * LROW + col8) = *(const h8*)(gB + row * DDIM + col8);
    }
#endif
  }
  __syncthreads();

  const _Float16* Bp = diagBlk ? As : Bs;

  // A fragment (16x32 f16): lane l15 holds row M=l15; lanes 0-15 K=0..15,
  // lanes 16-31 K=16..31. B fragment (32x16): lanes 0-15 K=0..15 of col N,
  // lanes 16-31 K=16..31. All reads from padded LDS rows (conflict-free).
  const _Float16* aRow = As + (w * 16 + l15) * LROW + hi * 16;
  const _Float16* bRow[4];
#pragma unroll
  for (int n = 0; n < 4; ++n)
    bRow[n] = Bp + (n * 16 + l15) * LROW + hi * 16;

  v8f acc[4] = {};

#pragma unroll
  for (int kk = 0; kk < DDIM; kk += 32) {
    h8 alo = *(const h8*)(aRow + kk);
    h8 ahi = *(const h8*)(aRow + kk + 8);
    v16h a = __builtin_shufflevector(alo, ahi, 0, 1, 2, 3, 4, 5, 6, 7,
                                     8, 9, 10, 11, 12, 13, 14, 15);
#pragma unroll
    for (int n = 0; n < 4; ++n) {
      h8 blo = *(const h8*)(bRow[n] + kk);
      h8 bhi = *(const h8*)(bRow[n] + kk + 8);
      v16h b = __builtin_shufflevector(blo, bhi, 0, 1, 2, 3, 4, 5, 6, 7,
                                       8, 9, 10, 11, 12, 13, 14, 15);
      acc[n] = __builtin_amdgcn_wmma_f32_16x16x32_f16(
          /*neg_a=*/false, a, /*neg_b=*/false, b,
          /*c_mod=*/(short)0, acc[n], /*reuse_a=*/false, /*reuse_b=*/false);
    }
  }

  // Epilogue: C/D layout — VGPR r, lanes 0-15: (M=r, N=lane); lanes 16-31: (M=8+r, N=lane-16)
  float si[8];
#pragma unroll
  for (int r = 0; r < 8; ++r) si[r] = sq[ibase + r + hi * 8];

  float accsum = 0.f;
#pragma unroll
  for (int n = 0; n < 4; ++n) {
    const int j    = jbase + n * 16 + l15;
    const float sj = sq[j];
#pragma unroll
    for (int r = 0; r < 8; ++r) {
      const int i = ibase + r + hi * 8;
      float l2 = si[r] + sj - 2.f * acc[n][r];
      l2 = fmaxf(l2, 0.f);
      // sum_{q=0..4} exp(-l2/(bw/4 * 2^q)) = t + sqrt(t) + ... (1 exp + 4 sqrt)
      float t0 = __expf(coef * l2);
      float ks = t0;
      float tt = t0;
#pragma unroll
      for (int q = 0; q < 4; ++q) { tt = __fsqrt_rn(tt); ks += tt; }
      const float sgn = (((i ^ j) & BHALF) ? -1.f : 1.f);  // xx,yy:+  xy,yx:-
      // symmetry weight: off-diag block counts (i,j) and (j,i); diag block:
      // strictly-upper x2, diagonal x1, strictly-lower x0.
      const float wgt = diagBlk ? ((j > i) ? 2.f : ((j == i) ? 1.f : 0.f)) : 2.f;
      accsum += wgt * sgn * ks;
    }
  }

  __shared__ float red[128];
  red[t] = accsum;
  __syncthreads();
  for (int st = 64; st > 0; st >>= 1) {
    if (t < st) red[t] += red[t + st];
    __syncthreads();
  }
  if (t == 0) part[blockIdx.x] = red[0];
}

// ---------------------------------------------------------------------------
// K4: final deterministic reduction of NPAIRS block partials -> mean
// ---------------------------------------------------------------------------
__global__ void __launch_bounds__(256)
mmd_finish_kernel(const char* __restrict__ ws, float* __restrict__ out) {
  const float* part = (const float*)(ws + PART_OFF);
  __shared__ float red[256];
  const int t = threadIdx.x;
  float s = 0.f;
  for (int i = t; i < NPAIRS; i += 256) s += part[i];
  red[t] = s;
  __syncthreads();
  for (int st = 128; st > 0; st >>= 1) {
    if (t < st) red[t] += red[t + st];
    __syncthreads();
  }
  if (t == 0) out[0] = red[0] / ((float)BHALF * (float)BHALF);
}

extern "C" void kernel_launch(void* const* d_in, const int* in_sizes, int n_in,
                              void* d_out, int out_size, void* d_ws, size_t ws_size,
                              hipStream_t stream) {
  (void)in_sizes; (void)n_in; (void)out_size; (void)ws_size;
  const float* src = (const float*)d_in[0];
  const float* tgt = (const float*)d_in[1];
  float* out = (float*)d_out;
  char* ws = (char*)d_ws;

  mmd_prep_kernel<<<256, 256, 0, stream>>>(src, tgt, ws);
  mmd_bw_kernel<<<1, 256, 0, stream>>>(ws);
  mmd_gemm_kernel<<<NPAIRS, 128, 0, stream>>>(ws);
  mmd_finish_kernel<<<1, 256, 0, stream>>>(ws, out);
}
